// CrystalEGNN_32753420599913
// MI455X (gfx1250) — matrix operand order repaired and checked by reading, not compiled, not run on hardware
//
#include <hip/hip_runtime.h>

#define B_    2
#define N_    320
#define D_    128
#define H_    128
#define L_    4
#define OH_   100
#define GATE_ 384

typedef __attribute__((ext_vector_type(16))) _Float16 v16h;
typedef __attribute__((ext_vector_type(8)))  _Float16 v8h;
typedef __attribute__((ext_vector_type(8)))  float    v8f;

// fast transcendentals: v_exp_f32 + v_rcp_f32 (TRANS pipe, co-executes with VALU)
__device__ __forceinline__ float silu_f(float v) {
    return v * __builtin_amdgcn_rcpf(1.0f + __expf(-v));
}
__device__ __forceinline__ float sigm_f(float v) {
    return __builtin_amdgcn_rcpf(1.0f + __expf(-v));
}
__device__ __forceinline__ float softplus_f(float v) {
    return v > 0.0f ? v + log1pf(expf(-v)) : log1pf(expf(v));   // 6 uses total: keep precise
}

// ---------------------------------------------------------------- 3x3 inverse
__global__ void k_invcell(const float* __restrict__ cell, float* __restrict__ invC) {
    int b = threadIdx.x;
    if (b >= B_) return;
    const float* c = cell + b * 9;
    float a = c[0], bb = c[1], cc = c[2];
    float d = c[3], e  = c[4], f  = c[5];
    float g = c[6], hh = c[7], ii = c[8];
    float A  = e * ii - f * hh;
    float Bv = f * g  - d * ii;
    float C  = d * hh - e * g;
    float det = a * A + bb * Bv + cc * C;
    float r = 1.0f / det;
    float* o = invC + b * 9;
    o[0] = A * r;  o[1] = (cc * hh - bb * ii) * r;  o[2] = (bb * f - cc * e) * r;
    o[3] = Bv * r; o[4] = (a * ii - cc * g)  * r;   o[5] = (cc * d - a * f) * r;
    o[6] = C * r;  o[7] = (bb * g - a * hh)  * r;   o[8] = (a * e - bb * d) * r;
}

// --------------------------------------------------------- input embedding MLP
__global__ __launch_bounds__(128) void k_embed(
    const float* __restrict__ pos, const float* __restrict__ onehot,
    const float* __restrict__ W1, const float* __restrict__ b1,
    const float* __restrict__ W2, const float* __restrict__ b2,
    float* __restrict__ hout)
{
    __shared__ float inb[OH_ + 3];
    __shared__ float mid[D_];
    const int row = blockIdx.x;
    const int tid = threadIdx.x;
    if (tid < OH_ + 3)
        inb[tid] = (tid < 3) ? pos[row * 3 + tid] : onehot[row * OH_ + tid - 3];
    __syncthreads();
    float t = b1[tid];
    for (int k = 0; k < OH_ + 3; ++k) t += inb[k] * W1[k * D_ + tid];
    mid[tid] = silu_f(t);
    __syncthreads();
    float o = b2[tid];
    for (int k = 0; k < D_; ++k) o += mid[k] * W2[k * D_ + tid];
    hout[row * D_ + tid] = o;
}

// ----------------------------------------------------- min-image rij/dij field
__global__ void k_minimage(const float* __restrict__ x, const float* __restrict__ mask,
                           const float* __restrict__ cell, const float* __restrict__ invC,
                           float* __restrict__ rij, float* __restrict__ dij)
{
    int idx = blockIdx.x * blockDim.x + threadIdx.x;
    if (idx >= B_ * N_ * N_) return;
    int b = idx / (N_ * N_);
    int r = idx % (N_ * N_);
    int i = r / N_, j = r % N_;
    const float* ic = invC + b * 9;
    const float* cm = cell + b * 9;
    const float* xi = x + (b * N_ + i) * 3;
    const float* xj = x + (b * N_ + j) * 3;
    float df[3];
#pragma unroll
    for (int m = 0; m < 3; ++m) {
        float fi = xi[0] * ic[m] + xi[1] * ic[3 + m] + xi[2] * ic[6 + m];
        float fj = xj[0] * ic[m] + xj[1] * ic[3 + m] + xj[2] * ic[6 + m];
        float d = fj - fi;
        df[m] = d - rintf(d);          // jnp.round == round half to even == rintf
    }
    float rr[3];
#pragma unroll
    for (int m = 0; m < 3; ++m)
        rr[m] = df[0] * cm[m] + df[1] * cm[3 + m] + df[2] * cm[6 + m];
    float pm = mask[b * N_ + i] * mask[b * N_ + j];
    float dd = sqrtf(rr[0] * rr[0] + rr[1] * rr[1] + rr[2] * rr[2] + 1e-12f);
    dd = dd * pm + (1.0f - pm) * 1e6f;
#pragma unroll
    for (int m = 0; m < 3; ++m) rij[idx * 3 + m] = rr[m] * pm;
    dij[idx] = dd;
}

// --------------------------------------- transpose+convert We2[l] to f16 [n][k]
__global__ void k_weprep(const float* __restrict__ We2l, _Float16* __restrict__ W2t) {
    const int n = blockIdx.x, k = threadIdx.x;
    W2t[n * H_ + k] = (_Float16)We2l[k * H_ + n];
}

// ------------------------------- per-node projections through the concat split
__global__ __launch_bounds__(128) void k_nodeproj(
    const float* __restrict__ h, const float* __restrict__ We1l,
    const float* __restrict__ be1l,
    float* __restrict__ Ah, float* __restrict__ Bh)
{
    __shared__ float hr[D_];
    const int row = blockIdx.x, tid = threadIdx.x;
    hr[tid] = h[row * D_ + tid];
    __syncthreads();
    float a = be1l[tid], b = 0.0f;
    for (int k = 0; k < D_; ++k) {
        const float hv = hr[k];
        a += hv * We1l[k * D_ + tid];          // rows 0..D-1   (h_i side)
        b += hv * We1l[(D_ + k) * D_ + tid];   // rows D..2D-1  (h_j side)
    }
    Ah[row * D_ + tid] = a;
    Bh[row * D_ + tid] = b;
}

// -------------------- fused per-node edge pipeline + GRU (the WMMA hot kernel)
__global__ __launch_bounds__(128) void k_edge(
    const float* __restrict__ h, const float* __restrict__ x,
    const float* __restrict__ mask,
    const float* __restrict__ rij, const float* __restrict__ dij,
    const float* __restrict__ Ah, const float* __restrict__ Bh,
    const _Float16* __restrict__ W2t,
    const float* __restrict__ wlast, const float* __restrict__ be2,
    const float* __restrict__ We3, const float* __restrict__ be3,
    const float* __restrict__ Wm1, const float* __restrict__ bm1,
    const float* __restrict__ Wm2, const float* __restrict__ bm2,
    const float* __restrict__ Wp1, const float* __restrict__ bp1,
    const float* __restrict__ Wp2, const float* __restrict__ bp2,
    const float* __restrict__ Wih, const float* __restrict__ Whh,
    const float* __restrict__ bih, const float* __restrict__ bhh,
    float* __restrict__ hout, float* __restrict__ xout)
{
    __shared__ __align__(16) _Float16 e1h[16][136];   // 16 edges x 128 K, f16
    __shared__ float m2s[16][132];                    // post-GEMM tile, f32
    __shared__ float hvec[D_], Ssum[D_], mnode[D_];
    __shared__ float sWe3[D_], sWp1[D_], sBp1[D_], sWp2[D_];
    __shared__ float muS[16], phiS[16], dloc[16], pmloc[16];
    __shared__ float rloc[16][3];
    __shared__ float red[16][8];
    __shared__ float xacc[3];

    const int tid  = threadIdx.x;
    const int b    = blockIdx.x / N_;
    const int i    = blockIdx.x % N_;
    const int node = b * N_ + i;
    const float maskI = mask[node];

    // per-thread loop invariants in registers
    const float air  = Ah[node * D_ + tid];
    const float wlr  = wlast[tid];
    const float wm1r = Wm1[tid];
    const float bm1r = bm1[tid];

    // small per-edge weight vectors staged in LDS once
    hvec[tid] = h[node * D_ + tid];
    Ssum[tid] = 0.0f;
    sWe3[tid] = We3[tid];
    sWp1[tid] = Wp1[tid];
    sBp1[tid] = bp1[tid];
    sWp2[tid] = Wp2[tid];
    if (tid < 3) xacc[tid] = 0.0f;

    const int lane = tid & 31;
    const int ln15 = lane & 15;
    const int hb   = lane >> 4;   // lane bank (K sub-block select)
    const int wv   = tid >> 5;    // wave id 0..3 -> output column block
    const float be2a = be2[wv * 32 + ln15];
    const float be2b = be2[wv * 32 + 16 + ln15];

    for (int jt = 0; jt < N_ / 16; ++jt) {
        const int j0 = jt * 16;
        if (tid < 16) {
            const int j = j0 + tid;
            const int eidx = node * N_ + j;
            dloc[tid]  = dij[eidx];
            pmloc[tid] = maskI * mask[b * N_ + j];
            rloc[tid][0] = rij[eidx * 3 + 0];
            rloc[tid][1] = rij[eidx * 3 + 1];
            rloc[tid][2] = rij[eidx * 3 + 2];
        }
        __syncthreads();

        // e1 = silu(Ah_i + Bh_j + dij * We1[2D]) -> f16 tile (16x128)
        {
            const int k = tid;
#pragma unroll 4
            for (int e = 0; e < 16; ++e) {
                const float bv = Bh[(b * N_ + j0 + e) * D_ + k];
                e1h[e][k] = (_Float16)silu_f(air + bv + dloc[e] * wlr);
            }
            if (jt + 1 < N_ / 16)
                __builtin_prefetch((const char*)&Bh[(b * N_ + j0 + 16) * D_] + tid * 64, 0, 3);
        }
        __syncthreads();

        // 16x128 = e1(16x128,f16) @ We2(128x128,f16), f32 accumulate via WMMA
#pragma unroll
        for (int nt = 0; nt < 2; ++nt) {
            const int n0 = wv * 32 + nt * 16;
            v8f acc = {};
#pragma unroll
            for (int c = 0; c < 4; ++c) {
                // A: row M=ln15; halves K = (c*32 + 8*hb)..+7 and +16..+23
                const _Float16* pa = &e1h[ln15][c * 32 + 8 * hb];
                v8h a0 = *(const v8h*)pa;
                v8h a1 = *(const v8h*)(pa + 16);
                v16h Am = __builtin_shufflevector(a0, a1,
                    0,1,2,3,4,5,6,7,8,9,10,11,12,13,14,15);
                // B: col n = n0+ln15; halves K = (c*32 + 8*hb)..+7 and +16..+23 (row of W2t)
                const _Float16* pb = &W2t[(n0 + ln15) * H_ + c * 32 + 8 * hb];
                v8h b0 = *(const v8h*)pb;
                v8h b1 = *(const v8h*)(pb + 16);
                v16h Bm = __builtin_shufflevector(b0, b1,
                    0,1,2,3,4,5,6,7,8,9,10,11,12,13,14,15);
                acc = __builtin_amdgcn_wmma_f32_16x16x32_f16(
                    false, Am, false, Bm, (short)0, acc, false, false);
            }
            const float bc = nt ? be2b : be2a;
            const int col = n0 + ln15;
#pragma unroll
            for (int r = 0; r < 8; ++r) {
                const int M = r + 8 * hb;
                m2s[M][col] = silu_f(acc[r] + bc);
            }
        }
        __syncthreads();

        // m_ij = m2 @ We3 + be3 (8 lanes per edge, then combine)
        {
            const int e = tid >> 3, k8 = tid & 7;
            float p = 0.0f;
#pragma unroll
            for (int q = 0; q < 16; ++q) {
                const int kk = k8 * 16 + q;
                p += m2s[e][kk] * sWe3[kk];
            }
            red[e][k8] = p;
        }
        __syncthreads();
        if (tid < 16) {
            float s = be3[0];
#pragma unroll
            for (int q = 0; q < 8; ++q) s += red[tid][q];
            muS[tid] = s * pmloc[tid];
        }
        __syncthreads();

        // phi(mu): scalar -> 128 hidden -> scalar
        {
            const int e = tid >> 3, k8 = tid & 7;
            const float t = muS[e];
            float p = 0.0f;
#pragma unroll
            for (int q = 0; q < 16; ++q) {
                const int kk = k8 * 16 + q;
                p += silu_f(t * sWp1[kk] + sBp1[kk]) * sWp2[kk];
            }
            red[e][k8] = p;
        }
        __syncthreads();
        if (tid < 16) {
            float s = bp2[0];
#pragma unroll
            for (int q = 0; q < 8; ++q) s += red[tid][q];
            phiS[tid] = s;
        }
        __syncthreads();

        // accumulate node-message hidden S[k] += sum_e silu(mu_e*Wm1+bm1)
        {
            float s = 0.0f;
#pragma unroll 4
            for (int e = 0; e < 16; ++e) s += silu_f(muS[e] * wm1r + bm1r);
            Ssum[tid] += s;
        }
        // accumulate coordinate update
        if (tid < 3) {
            float s = 0.0f;
#pragma unroll 4
            for (int e = 0; e < 16; ++e) s += rloc[e][tid] * phiS[e];
            xacc[tid] += s;
        }
        __syncthreads();
    }

    // m_node = S @ Wm2 + N*bm2   (sum over all N j's includes bias N times)
    {
        float mn = (float)N_ * bm2[tid];
        for (int k = 0; k < H_; ++k) mn += Ssum[k] * Wm2[k * D_ + tid];
        mnode[tid] = mn;
    }
    __syncthreads();

    // GRU cell (torch gate order r,z,n)
    {
        const int d = tid;
        float gir = bih[d], giz = bih[D_ + d], gin = bih[2 * D_ + d];
        float ghr = bhh[d], ghz = bhh[D_ + d], ghn = bhh[2 * D_ + d];
        for (int k = 0; k < D_; ++k) {
            const float mk = mnode[k], hk = hvec[k];
            const float* wi = &Wih[k * GATE_];
            const float* wh = &Whh[k * GATE_];
            gir += mk * wi[d];        giz += mk * wi[D_ + d];   gin += mk * wi[2 * D_ + d];
            ghr += hk * wh[d];        ghz += hk * wh[D_ + d];   ghn += hk * wh[2 * D_ + d];
        }
        const float rg = sigm_f(gir + ghr);
        const float zg = sigm_f(giz + ghz);
        const float ng = tanhf(gin + rg * ghn);
        hout[node * D_ + d] = ((1.0f - zg) * ng + zg * hvec[d]) * maskI;
    }
    if (tid < 3) xout[node * 3 + tid] = x[node * 3 + tid] + xacc[tid];
}

// ------------------------------------------------------------- pooled readout
__global__ __launch_bounds__(128) void k_readout(
    const float* __restrict__ h, const float* __restrict__ mask,
    const float* __restrict__ Wc1, const float* __restrict__ bc1,
    const float* __restrict__ Wc2, const float* __restrict__ bc2,
    const float* __restrict__ Wc3, const float* __restrict__ bc3,
    float* __restrict__ out)
{
    __shared__ float feat[D_], o1[D_], o2[64], dsh[1];
    const int tid = threadIdx.x;
    for (int b = 0; b < B_; ++b) {
        if (tid == 0) {
            float s = 0.0f;
            for (int i = 0; i < N_; ++i) s += mask[b * N_ + i];
            dsh[0] = fmaxf(s, 1.0f);
        }
        float s = 0.0f;
        for (int i = 0; i < N_; ++i) s += h[(b * N_ + i) * D_ + tid] * mask[b * N_ + i];
        __syncthreads();
        feat[tid] = s / dsh[0];
        __syncthreads();
        float t = bc1[tid];
        for (int k = 0; k < D_; ++k) t += feat[k] * Wc1[k * D_ + tid];
        o1[tid] = silu_f(t);
        __syncthreads();
        if (tid < 64) {
            float u = bc2[tid];
            for (int k = 0; k < D_; ++k) u += o1[k] * Wc2[k * 64 + tid];
            o2[tid] = silu_f(u);
        }
        __syncthreads();
        if (tid < 6) {
            float v = bc3[tid];
            for (int k = 0; k < 64; ++k) v += o2[k] * Wc3[k * 6 + tid];
            out[b * 6 + tid] = (tid < 3) ? softplus_f(v) : sigm_f(v) * 180.0f;
        }
        __syncthreads();
    }
}

// ============================================================================
extern "C" void kernel_launch(void* const* d_in, const int* in_sizes, int n_in,
                              void* d_out, int out_size, void* d_ws, size_t ws_size,
                              hipStream_t stream)
{
    (void)in_sizes; (void)n_in; (void)out_size; (void)ws_size;
    const float* pos    = (const float*)d_in[0];
    const float* onehot = (const float*)d_in[1];
    const float* mask   = (const float*)d_in[2];
    const float* cell   = (const float*)d_in[3];
    const float* W_in1  = (const float*)d_in[4];
    const float* b_in1  = (const float*)d_in[5];
    const float* W_in2  = (const float*)d_in[6];
    const float* b_in2  = (const float*)d_in[7];
    const float* We1 = (const float*)d_in[8];
    const float* be1 = (const float*)d_in[9];
    const float* We2 = (const float*)d_in[10];
    const float* be2 = (const float*)d_in[11];
    const float* We3 = (const float*)d_in[12];
    const float* be3 = (const float*)d_in[13];
    const float* Wm1 = (const float*)d_in[14];
    const float* bm1 = (const float*)d_in[15];
    const float* Wm2 = (const float*)d_in[16];
    const float* bm2 = (const float*)d_in[17];
    const float* Wp1 = (const float*)d_in[18];
    const float* bp1 = (const float*)d_in[19];
    const float* Wp2 = (const float*)d_in[20];
    const float* bp2 = (const float*)d_in[21];
    const float* W_ih = (const float*)d_in[22];
    const float* W_hh = (const float*)d_in[23];
    const float* b_ih = (const float*)d_in[24];
    const float* b_hh = (const float*)d_in[25];
    const float* Wc1 = (const float*)d_in[26];
    const float* bc1 = (const float*)d_in[27];
    const float* Wc2 = (const float*)d_in[28];
    const float* bc2 = (const float*)d_in[29];
    const float* Wc3 = (const float*)d_in[30];
    const float* bc3 = (const float*)d_in[31];

    char* wsp = (char*)d_ws;
    size_t off = 0;
    auto alloc = [&](size_t bytes) -> char* {
        char* p = wsp + off;
        off += (bytes + 255) & ~(size_t)255;
        return p;
    };
    float* invC = (float*)alloc(B_ * 9 * sizeof(float));
    float* hA   = (float*)alloc((size_t)B_ * N_ * D_ * sizeof(float));
    float* hB   = (float*)alloc((size_t)B_ * N_ * D_ * sizeof(float));
    float* xA   = (float*)alloc((size_t)B_ * N_ * 3 * sizeof(float));
    float* xB   = (float*)alloc((size_t)B_ * N_ * 3 * sizeof(float));
    float* rij  = (float*)alloc((size_t)B_ * N_ * N_ * 3 * sizeof(float));
    float* dij  = (float*)alloc((size_t)B_ * N_ * N_ * sizeof(float));
    float* Ahb  = (float*)alloc((size_t)B_ * N_ * D_ * sizeof(float));
    float* Bhb  = (float*)alloc((size_t)B_ * N_ * D_ * sizeof(float));
    _Float16* W2t = (_Float16*)alloc((size_t)H_ * H_ * sizeof(_Float16));

    const int ne = B_ * N_ * N_;

    k_invcell<<<1, 32, 0, stream>>>(cell, invC);
    k_embed<<<B_ * N_, 128, 0, stream>>>(pos, onehot, W_in1, b_in1, W_in2, b_in2, hA);
    k_minimage<<<(ne + 255) / 256, 256, 0, stream>>>(pos, mask, cell, invC, rij, dij);

    float* hbuf[2] = { hA, hB };
    float* xbuf[2] = { xA, xB };
    const float* hcur = hA;
    const float* xcur = pos;

    for (int l = 0; l < L_; ++l) {
        const float* We1l = We1 + (size_t)l * (2 * D_ + 1) * D_;
        float* hdst = hbuf[(l + 1) & 1];
        float* xdst = xbuf[l & 1];

        k_weprep<<<H_, H_, 0, stream>>>(We2 + (size_t)l * H_ * H_, W2t);
        k_nodeproj<<<B_ * N_, 128, 0, stream>>>(hcur, We1l, be1 + l * D_, Ahb, Bhb);
        k_edge<<<B_ * N_, 128, 0, stream>>>(
            hcur, xcur, mask, rij, dij, Ahb, Bhb, W2t,
            We1l + (size_t)2 * D_ * D_,      // We1[2D] row (dij weights)
            be2 + l * H_,
            We3 + l * H_, be3 + l,
            Wm1 + l * H_, bm1 + l * H_,
            Wm2 + (size_t)l * H_ * D_, bm2 + l * D_,
            Wp1 + l * H_, bp1 + l * H_, Wp2 + l * H_, bp2 + l,
            W_ih + (size_t)l * D_ * GATE_, W_hh + (size_t)l * D_ * GATE_,
            b_ih + l * GATE_, b_hh + l * GATE_,
            hdst, xdst);
        if (l < L_ - 1)
            k_minimage<<<(ne + 255) / 256, 256, 0, stream>>>(xdst, mask, cell, invC, rij, dij);
        hcur = hdst;
        xcur = xdst;
    }

    k_readout<<<1, 128, 0, stream>>>(hcur, mask, Wc1, bc1, Wc2, bc2, Wc3, bc3,
                                     (float*)d_out);
}
